// Encoder_21646635171889
// MI455X (gfx1250) — matrix operand order repaired and checked by reading, not compiled, not run on hardware
//
#include <hip/hip_runtime.h>
#include <hip/hip_bf16.h>
#include <math.h>

#define N_NODES 50000
#define N_EDGES 800000
#define FIN     128
#define DDIM    64
#define DD0     256
#define NGRAPH  256
#define OUTW    (DD0 + DDIM)   /* 320 */

typedef __bf16 bf16;
typedef __attribute__((ext_vector_type(8)))  bf16  v8bf;
typedef __attribute__((ext_vector_type(16))) bf16  v16bf;
typedef __attribute__((ext_vector_type(8)))  float v8f;

// ---------- orderable-uint encoding for float atomicMax ----------
__device__ __forceinline__ unsigned f2ord(float f) {
    unsigned u = __float_as_uint(f);
    return (u & 0x80000000u) ? ~u : (u | 0x80000000u);
}
__device__ __forceinline__ float ord2f(unsigned u) {
    unsigned v = (u & 0x80000000u) ? (u & 0x7FFFFFFFu) : ~u;
    return __uint_as_float(v);
}

// ---------- small utility kernels ----------
__global__ void fill_f32_kernel(float* p, float v, int n) {
    int i = blockIdx.x * blockDim.x + threadIdx.x;
    if (i < n) p[i] = v;
}
__global__ void fill_u32_kernel(unsigned* p, unsigned v, int n) {
    int i = blockIdx.x * blockDim.x + threadIdx.x;
    if (i < n) p[i] = v;
}
__global__ void copy_f32_kernel(const float* __restrict__ s, float* __restrict__ d, int n) {
    int i = blockIdx.x * blockDim.x + threadIdx.x;
    if (i < n) d[i] = s[i];
}
__global__ void cast_bf16_kernel(const float* __restrict__ s, bf16* __restrict__ d, int n) {
    int i = blockIdx.x * blockDim.x + threadIdx.x;
    if (i < n) d[i] = (bf16)s[i];
}
// W [K x Nc] fp32 (row-major) -> Wt rows [rowOff, rowOff+Nc), Wt[(rowOff+n)*K + k], bf16
__global__ void transpose_cast_kernel(const float* __restrict__ W, bf16* __restrict__ Wt,
                                      int K, int Nc, int rowOff) {
    int idx = blockIdx.x * blockDim.x + threadIdx.x;
    if (idx >= K * Nc) return;
    int k = idx / Nc, n = idx % Nc;          // coalesced read of W
    Wt[(size_t)(rowOff + n) * K + k] = (bf16)W[idx];
}

// ---------- WMMA fragment load helper ----------
__device__ __forceinline__ v16bf load_frag(const bf16* __restrict__ p0,
                                           const bf16* __restrict__ p1) {
    v8bf x = *(const v8bf*)p0;
    v8bf y = *(const v8bf*)p1;
    return __builtin_shufflevector(x, y, 0,1,2,3,4,5,6,7,8,9,10,11,12,13,14,15);
}

// ---------- WMMA bf16 GEMM:  C[M,NC] = A[M,K] @ Wt[NC,K]^T + bias ----------
// block = 128 threads (4 waves). Each wave computes a 16(M) x 64(N) slab with
// 4 f32 accumulators: one A fragment load feeds 4 WMMAs (4x less A refetch
// from L2, 4x higher WMMA density per loop overhead). All four B fragments
// are materialized before the WMMA sequence so the loads issue as one clause
// and the per-WMMA loadcnt waits can be partial (load/WMMA overlap).
// wave w handles columns [ (blockIdx.y*4 + w)*64 , +64 )
__global__ void wmma_gemm_bf16_kernel(const bf16* __restrict__ A, const bf16* __restrict__ Wt,
                                      const float* __restrict__ bias, float* __restrict__ C,
                                      int M, int K, int NC, int ldc) {
    const int wave = threadIdx.x >> 5;
    const int lane = threadIdx.x & 31;
    const int hi   = lane >> 4;          // 0 / 1 (lane half)
    const int lm   = lane & 15;
    const int mBase = blockIdx.x * 16;
    const int nSlab = (blockIdx.y * 4 + wave) * 64;
    if (mBase >= M || nSlab >= NC) return;

    const bf16* Arow  = A  + (size_t)(mBase + lm) * K;   // row M = mBase+lm
    const bf16* Brow0 = Wt + (size_t)(nSlab + lm) * K;   // col N = nSlab+lm (+t*16 rows)

    v8f cacc[4] = {{}, {}, {}, {}};
    for (int k0 = 0; k0 < K; k0 += 32) {
        // A fragment: elems 0..7 -> K = k0 + hi*8 + i ; elems 8..15 -> +16
        v16bf a = load_frag(Arow + k0 + hi * 8, Arow + k0 + 16 + hi * 8);
        const int kb = k0 + hi * 8;      // B fragment: same K order as A

        v16bf b[4];
#pragma unroll
        for (int t = 0; t < 4; ++t) {
            const bf16* bp = Brow0 + (size_t)t * 16 * K + kb;
            b[t] = load_frag(bp, bp + 16);
        }
        if (k0 + 32 < K) {
            __builtin_prefetch(Arow + k0 + 32, 0, 1);    // global_prefetch_b8
            __builtin_prefetch(Brow0 + k0 + 32, 0, 1);
        }
#pragma unroll
        for (int t = 0; t < 4; ++t)
            cacc[t] = __builtin_amdgcn_wmma_f32_16x16x32_bf16(
                false, a, false, b[t], (short)0, cacc[t], false, false);
    }

#pragma unroll
    for (int t = 0; t < 4; ++t) {
        const int n  = nSlab + t * 16 + lm;
        const float bv = bias ? bias[n] : 0.0f;
        float* Crow = C + (size_t)(mBase + hi * 8) * ldc + n;  // VGPR r -> row mBase+hi*8+r
#pragma unroll
        for (int r = 0; r < 8; ++r)
            Crow[(size_t)r * ldc] = cacc[t][r] + bv;
    }
}

// ---------- edge attention kernels ----------
// one wave per edge: logits[e] = scale * dot(q[dst], k[src] + a*We) ; atomicMax m[dst]
__global__ void edge_logits_kernel(const float* __restrict__ C, int ldc, int qOff, int kOff, int D,
                                   const float* __restrict__ We, const int* __restrict__ ei,
                                   const float* __restrict__ ea, float* __restrict__ logits,
                                   unsigned* __restrict__ menc, float scale, int E) {
    int e    = blockIdx.x * (blockDim.x >> 5) + (threadIdx.x >> 5);
    int lane = threadIdx.x & 31;
    if (e >= E) return;
    int src = ei[e], dst = ei[E + e];
    float a = ea[e];
    const float* qr = C + (size_t)dst * ldc + qOff;
    const float* kr = C + (size_t)src * ldc + kOff;
    float acc = 0.0f;
    for (int d = lane; d < D; d += 32) acc += qr[d] * (kr[d] + a * We[d]);
#pragma unroll
    for (int off = 16; off; off >>= 1) acc += __shfl_xor(acc, off, 32);
    if (lane == 0) {
        float lg = acc * scale;
        logits[e] = lg;
        atomicMax(&menc[dst], f2ord(lg));
    }
}

__global__ void edge_exp_kernel(const float* __restrict__ logits, const unsigned* __restrict__ menc,
                                const int* __restrict__ ei, float* __restrict__ p,
                                float* __restrict__ s, int E) {
    int e = blockIdx.x * blockDim.x + threadIdx.x;
    if (e >= E) return;
    int dst = ei[E + e];
    float pe = __expf(logits[e] - ord2f(menc[dst]));
    p[e] = pe;
    atomicAdd(&s[dst], pe);
}

// one wave per edge: out[dst] += (v[src] + a*We) * alpha   (out slice pre-loaded with skip GEMM)
__global__ void edge_msg_kernel(float* __restrict__ C, int ldc, int vOff, int outOff, int D,
                                const float* __restrict__ We, const int* __restrict__ ei,
                                const float* __restrict__ ea, const float* __restrict__ p,
                                const float* __restrict__ s, int E) {
    int e    = blockIdx.x * (blockDim.x >> 5) + (threadIdx.x >> 5);
    int lane = threadIdx.x & 31;
    if (e >= E) return;
    int src = ei[e], dst = ei[E + e];
    float a = ea[e];
    float alpha = p[e] / (s[dst] + 1e-16f);
    const float* vr = C + (size_t)src * ldc + vOff;
    float* outr     = C + (size_t)dst * ldc + outOff;
    for (int d = lane; d < D; d += 32)
        atomicAdd(&outr[d], (vr[d] + a * We[d]) * alpha);
}

// ---------- batch norm ----------
__global__ void bn_stats_kernel(const float* __restrict__ X, int ldx, int colOff, int Nrows,
                                float* __restrict__ mu, float* __restrict__ rsig) {
    int col = blockIdx.x;
    __shared__ float sh_s[256], sh_q[256];
    float s = 0.0f, q = 0.0f;
    for (int i = threadIdx.x; i < Nrows; i += blockDim.x) {
        float v = X[(size_t)i * ldx + colOff + col];
        s += v; q += v * v;
    }
    sh_s[threadIdx.x] = s; sh_q[threadIdx.x] = q;
    __syncthreads();
    for (int off = blockDim.x >> 1; off > 0; off >>= 1) {
        if ((int)threadIdx.x < off) {
            sh_s[threadIdx.x] += sh_s[threadIdx.x + off];
            sh_q[threadIdx.x] += sh_q[threadIdx.x + off];
        }
        __syncthreads();
    }
    if (threadIdx.x == 0) {
        float m   = sh_s[0] / (float)Nrows;
        float var = sh_q[0] / (float)Nrows - m * m;
        mu[col]   = m;
        rsig[col] = rsqrtf(var + 1e-5f);
    }
}

__global__ void bn_apply_kernel(const float* __restrict__ X, int ldx, int colOff,
                                const float* __restrict__ mu, const float* __restrict__ rsig,
                                const float* __restrict__ gamma, const float* __restrict__ beta,
                                int Nrows, int D, float* __restrict__ Y, int ldy, int ycolOff,
                                bf16* __restrict__ Ybf, int ldyb) {
    long long idx = (long long)blockIdx.x * blockDim.x + threadIdx.x;
    if (idx >= (long long)Nrows * D) return;
    int col = (int)(idx % D);
    int row = (int)(idx / D);
    float v = X[(size_t)row * ldx + colOff + col];
    float y = gamma[col] * (v - mu[col]) * rsig[col] + beta[col];
    Y[(size_t)row * ldy + ycolOff + col] = y;
    if (Ybf) Ybf[(size_t)row * ldyb + col] = (bf16)y;
}

// ---------- global mean pool ----------
__global__ void pool_accum_kernel(const float* __restrict__ xs, const int* __restrict__ batch,
                                  float* __restrict__ psum, float* __restrict__ pcnt, int Nrows) {
    int row = blockIdx.x;            // one block (OUTW threads) per node
    int c   = threadIdx.x;
    if (row >= Nrows) return;
    int g = batch[row];
    atomicAdd(&psum[(size_t)g * OUTW + c], xs[(size_t)row * OUTW + c]);
    if (c == 0) atomicAdd(&pcnt[g], 1.0f);
}
__global__ void pool_final_kernel(const float* __restrict__ psum, const float* __restrict__ pcnt,
                                  float* __restrict__ out, int n) {
    int i = blockIdx.x * blockDim.x + threadIdx.x;
    if (i >= n) return;
    out[i] = psum[i] / fmaxf(pcnt[i / OUTW], 1.0f);
}

// =====================================================================
extern "C" void kernel_launch(void* const* d_in, const int* in_sizes, int n_in,
                              void* d_out, int out_size, void* d_ws, size_t ws_size,
                              hipStream_t stream) {
    (void)in_sizes; (void)n_in; (void)out_size; (void)ws_size;
    // ---- inputs (setup_inputs dict order) ----
    const float* x      = (const float*)d_in[0];
    const int*   ei     = (const int*)  d_in[1];   // [2,E] flat: [0..E)=src, [E..2E)=dst
    const float* ea     = (const float*)d_in[2];
    const int*   batch  = (const int*)  d_in[3];
    const float* Wq0    = (const float*)d_in[4];
    const float* bq0    = (const float*)d_in[5];
    const float* Wk0    = (const float*)d_in[6];
    const float* bk0    = (const float*)d_in[7];
    const float* Wv0    = (const float*)d_in[8];
    const float* bv0    = (const float*)d_in[9];
    const float* We0    = (const float*)d_in[10];
    const float* Ws0    = (const float*)d_in[11];
    const float* bs0    = (const float*)d_in[12];
    const float* gamma0 = (const float*)d_in[13];
    const float* beta0  = (const float*)d_in[14];
    const float* Wq1    = (const float*)d_in[15];
    const float* bq1    = (const float*)d_in[16];
    const float* Wk1    = (const float*)d_in[17];
    const float* bk1    = (const float*)d_in[18];
    const float* Wv1    = (const float*)d_in[19];
    const float* bv1    = (const float*)d_in[20];
    const float* We1    = (const float*)d_in[21];
    const float* Ws1    = (const float*)d_in[22];
    const float* bs1    = (const float*)d_in[23];
    const float* gamma1 = (const float*)d_in[24];
    const float* beta1  = (const float*)d_in[25];

    // ---- workspace carve (256B aligned) ----
    char* w = (char*)d_ws;
    auto take = [&](size_t bytes) -> void* {
        void* p = (void*)w;
        w += (bytes + 255) & ~(size_t)255;
        return p;
    };
    bf16*     xb    = (bf16*)    take((size_t)N_NODES * FIN * sizeof(bf16));
    bf16*     Wt0   = (bf16*)    take((size_t)4 * DD0 * FIN * sizeof(bf16));
    float*    bias0 = (float*)   take((size_t)4 * DD0 * sizeof(float));
    float*    C0    = (float*)   take((size_t)N_NODES * 4 * DD0 * sizeof(float));
    bf16*     h0b   = (bf16*)    take((size_t)N_NODES * DD0 * sizeof(bf16));
    bf16*     Wt1   = (bf16*)    take((size_t)4 * DDIM * DD0 * sizeof(bf16));
    float*    bias1 = (float*)   take((size_t)4 * DDIM * sizeof(float));
    float*    C1    = C0;                               // layer-0 qkv dead by then: alias
    float*    lg    = (float*)   take((size_t)N_EDGES * sizeof(float));
    float*    pe    = (float*)   take((size_t)N_EDGES * sizeof(float));
    unsigned* menc  = (unsigned*)take((size_t)N_NODES * sizeof(unsigned));
    float*    sden  = (float*)   take((size_t)N_NODES * sizeof(float));
    float*    mu    = (float*)   take((size_t)DD0 * sizeof(float));
    float*    rsig  = (float*)   take((size_t)DD0 * sizeof(float));
    float*    psum  = (float*)   take((size_t)NGRAPH * OUTW * sizeof(float));
    float*    pcnt  = (float*)   take((size_t)NGRAPH * sizeof(float));

    float* xpool = (float*)d_out;                    // [G, 320]
    float* xs    = xpool + (size_t)NGRAPH * OUTW;    // [N, 320]

    const unsigned NEG_INF_ORD = 0x007FFFFFu;        // f2ord(-inf)
    #define CD(n, b) dim3(((n) + (b) - 1) / (b))

    // ---- cast x -> bf16 ----
    cast_bf16_kernel<<<CD(N_NODES * FIN, 256), 256, 0, stream>>>(x, xb, N_NODES * FIN);

    // =============== layer 0 ===============
    {
        const int K = FIN, Nc = DD0, NC = 4 * DD0;
        transpose_cast_kernel<<<CD(K * Nc, 256), 256, 0, stream>>>(Wq0, Wt0, K, Nc, 0);
        transpose_cast_kernel<<<CD(K * Nc, 256), 256, 0, stream>>>(Wk0, Wt0, K, Nc, Nc);
        transpose_cast_kernel<<<CD(K * Nc, 256), 256, 0, stream>>>(Wv0, Wt0, K, Nc, 2 * Nc);
        transpose_cast_kernel<<<CD(K * Nc, 256), 256, 0, stream>>>(Ws0, Wt0, K, Nc, 3 * Nc);
        copy_f32_kernel<<<CD(Nc, 256), 256, 0, stream>>>(bq0, bias0,          Nc);
        copy_f32_kernel<<<CD(Nc, 256), 256, 0, stream>>>(bk0, bias0 + Nc,     Nc);
        copy_f32_kernel<<<CD(Nc, 256), 256, 0, stream>>>(bv0, bias0 + 2 * Nc, Nc);
        copy_f32_kernel<<<CD(Nc, 256), 256, 0, stream>>>(bs0, bias0 + 3 * Nc, Nc);

        wmma_gemm_bf16_kernel<<<dim3(N_NODES / 16, NC / 256), 128, 0, stream>>>(
            xb, Wt0, bias0, C0, N_NODES, K, NC, NC);

        fill_u32_kernel<<<CD(N_NODES, 256), 256, 0, stream>>>(menc, NEG_INF_ORD, N_NODES);
        fill_f32_kernel<<<CD(N_NODES, 256), 256, 0, stream>>>(sden, 0.0f, N_NODES);

        float scale = 1.0f / sqrtf((float)Nc);
        edge_logits_kernel<<<CD(N_EDGES, 8), 256, 0, stream>>>(
            C0, NC, 0, Nc, Nc, We0, ei, ea, lg, menc, scale, N_EDGES);
        edge_exp_kernel<<<CD(N_EDGES, 256), 256, 0, stream>>>(lg, menc, ei, pe, sden, N_EDGES);
        edge_msg_kernel<<<CD(N_EDGES, 8), 256, 0, stream>>>(
            C0, NC, 2 * Nc, 3 * Nc, Nc, We0, ei, ea, pe, sden, N_EDGES);

        bn_stats_kernel<<<Nc, 256, 0, stream>>>(C0, NC, 3 * Nc, N_NODES, mu, rsig);
        bn_apply_kernel<<<CD((long long)N_NODES * Nc, 256), 256, 0, stream>>>(
            C0, NC, 3 * Nc, mu, rsig, gamma0, beta0, N_NODES, Nc,
            xs, OUTW, 0, h0b, Nc);   // writes xs cols [0,256) + bf16 copy for layer 1
    }

    // =============== layer 1 ===============
    {
        const int K = DD0, Nc = DDIM, NC = 4 * DDIM;
        transpose_cast_kernel<<<CD(K * Nc, 256), 256, 0, stream>>>(Wq1, Wt1, K, Nc, 0);
        transpose_cast_kernel<<<CD(K * Nc, 256), 256, 0, stream>>>(Wk1, Wt1, K, Nc, Nc);
        transpose_cast_kernel<<<CD(K * Nc, 256), 256, 0, stream>>>(Wv1, Wt1, K, Nc, 2 * Nc);
        transpose_cast_kernel<<<CD(K * Nc, 256), 256, 0, stream>>>(Ws1, Wt1, K, Nc, 3 * Nc);
        copy_f32_kernel<<<CD(Nc, 256), 256, 0, stream>>>(bq1, bias1,          Nc);
        copy_f32_kernel<<<CD(Nc, 256), 256, 0, stream>>>(bk1, bias1 + Nc,     Nc);
        copy_f32_kernel<<<CD(Nc, 256), 256, 0, stream>>>(bv1, bias1 + 2 * Nc, Nc);
        copy_f32_kernel<<<CD(Nc, 256), 256, 0, stream>>>(bs1, bias1 + 3 * Nc, Nc);

        wmma_gemm_bf16_kernel<<<dim3(N_NODES / 16, NC / 256), 128, 0, stream>>>(
            h0b, Wt1, bias1, C1, N_NODES, K, NC, NC);

        fill_u32_kernel<<<CD(N_NODES, 256), 256, 0, stream>>>(menc, NEG_INF_ORD, N_NODES);
        fill_f32_kernel<<<CD(N_NODES, 256), 256, 0, stream>>>(sden, 0.0f, N_NODES);

        float scale = 1.0f / sqrtf((float)Nc);
        edge_logits_kernel<<<CD(N_EDGES, 8), 256, 0, stream>>>(
            C1, NC, 0, Nc, Nc, We1, ei, ea, lg, menc, scale, N_EDGES);
        edge_exp_kernel<<<CD(N_EDGES, 256), 256, 0, stream>>>(lg, menc, ei, pe, sden, N_EDGES);
        edge_msg_kernel<<<CD(N_EDGES, 8), 256, 0, stream>>>(
            C1, NC, 2 * Nc, 3 * Nc, Nc, We1, ei, ea, pe, sden, N_EDGES);

        bn_stats_kernel<<<Nc, 256, 0, stream>>>(C1, NC, 3 * Nc, N_NODES, mu, rsig);
        bn_apply_kernel<<<CD((long long)N_NODES * Nc, 256), 256, 0, stream>>>(
            C1, NC, 3 * Nc, mu, rsig, gamma1, beta1, N_NODES, Nc,
            xs, OUTW, DD0, (bf16*)nullptr, 0);   // writes xs cols [256,320)
    }

    // =============== global mean pool -> xpool ===============
    fill_f32_kernel<<<CD(NGRAPH * OUTW, 256), 256, 0, stream>>>(psum, 0.0f, NGRAPH * OUTW);
    fill_f32_kernel<<<CD(NGRAPH, 256), 256, 0, stream>>>(pcnt, 0.0f, NGRAPH);
    pool_accum_kernel<<<N_NODES, OUTW, 0, stream>>>(xs, batch, psum, pcnt, N_NODES);
    pool_final_kernel<<<CD(NGRAPH * OUTW, 256), 256, 0, stream>>>(psum, pcnt, xpool, NGRAPH * OUTW);

    #undef CD
}
